// Matcher_57226144252718
// MI455X (gfx1250) — compile-verified, matrix operand order checked by reading, not run on hardware
//
#include <hip/hip_runtime.h>

// Problem constants (from reference)
#define BB   32
#define NN   4096
#define CC   558
#define MM   559      // CC + 1 (dustbin)
#define MP   560      // padded (multiple of 4 for b128 staging)
#define ITERS 5

typedef __attribute__((ext_vector_type(16))) _Float16 v16h;
typedef __attribute__((ext_vector_type(8)))  float    v8f;
typedef __attribute__((ext_vector_type(4)))  int      v4i;

#define NEG_INF (-__builtin_inff())
// log_nu = log(1/559)
#define LOG_NU  (-6.326149473f)

#if defined(__AMDGCN__) && __has_builtin(__builtin_amdgcn_global_load_async_to_lds_b128) && __has_builtin(__builtin_amdgcn_s_wait_asynccnt)
#define HAVE_ASYNC 1
#else
#define HAVE_ASYNC 0
#endif

// Stage n floats (n % 4 == 0, 16B-aligned src/dst) from global into LDS.
// Uses gfx1250 GLOBAL_LOAD_ASYNC_TO_LDS_B128 (ASYNCcnt) when available.
// Caller must __syncthreads() afterwards.
__device__ __forceinline__ void stage_lds(float* dst, const float* src,
                                          int n, int t, int nt)
{
#if HAVE_ASYNC
    for (int i = t * 4; i < n; i += nt * 4)
        __builtin_amdgcn_global_load_async_to_lds_b128(
            (__attribute__((address_space(1))) v4i*)(src + i),
            (__attribute__((address_space(3))) v4i*)(dst + i),
            0, 0);
    __builtin_amdgcn_s_wait_asynccnt(0);
#else
    for (int i = t; i < n; i += nt) dst[i] = src[i];
#endif
}

// ---------------------------------------------------------------------------
// K0: per-batch init: num_vis -> log_mu; zero log_v
// ---------------------------------------------------------------------------
__global__ void k_init(const int* __restrict__ mask,
                       float* __restrict__ logmu,
                       float* __restrict__ logv)
{
    int b = blockIdx.x;
    int t = threadIdx.x;
    __shared__ int red[256];
    int cnt = 0;
    for (int i = t; i < NN; i += 256) cnt += (mask[b * NN + i] != 0) ? 1 : 0;
    red[t] = cnt;
    __syncthreads();
    for (int s = 128; s > 0; s >>= 1) {
        if (t < s) red[t] += red[t + s];
        __syncthreads();
    }
    float lognv = __logf((float)red[0]);
    for (int i = t; i < NN; i += 256)
        logmu[b * NN + i] = (mask[b * NN + i] != 0) ? -lognv : NEG_INF;
    for (int i = t; i < MP; i += 256)
        logv[b * MP + i] = 0.0f;
}

// ---------------------------------------------------------------------------
// K1: row update.  log_u[b,n] = log_mu[b,n] - log( sum_m exp(aug[b,n,m]+log_v[b,m]) )
// One wave handles 16 rows; exp terms packed into WMMA f16 A-layout and
// reduced with V_WMMA_F32_16X16X32_F16 against an all-ones B (f32 accum).
// Main loop (cols 0..543) is divergence-free; tail chunk is branchless selects.
// ---------------------------------------------------------------------------
__global__ void k_row(const float* __restrict__ scores,
                      const float* __restrict__ dustp,
                      const float* __restrict__ logmu,
                      const float* __restrict__ logv,
                      float* __restrict__ logu)
{
    int b    = blockIdx.x >> 5;        // 32 row-tiles per batch
    int tile = blockIdx.x & 31;
    int t    = threadIdx.x;
    int wave = t >> 5;
    int lane = t & 31;
    int n0   = tile * 128 + wave * 16; // 8 waves * 16 rows = 128 rows/block

    __shared__ __attribute__((aligned(16))) float s_lv[MP];
    stage_lds(s_lv, logv + (size_t)b * MP, MP, t, 256);
    __syncthreads();

    float dust = dustp[0];
    int   row  = n0 + (lane & 15);
    const float* srow = scores + ((size_t)b * NN + row) * CC;
    int   hi   = (lane >> 4) * 8;  // lanes 0-15: K 0-7/16-23 ; lanes 16-31: K 8-15/24-31

    v8f acc = {};
    v16h ones;
#pragma unroll
    for (int j = 0; j < 16; j++) ones[j] = (_Float16)1.0f;

    // 17 full chunks: every lane's cols are < 544 < CC -- no bounds checks.
    for (int c0 = 0; c0 < 544; c0 += 32) {
        const float* p = srow + c0 + hi;
        const float* q = s_lv + c0 + hi;
        v16h a;
#pragma unroll
        for (int j = 0; j < 8; j++) {
            a[j]     = (_Float16)__expf(p[j]      + q[j]);
            a[8 + j] = (_Float16)__expf(p[16 + j] + q[16 + j]);
        }
        acc = __builtin_amdgcn_wmma_f32_16x16x32_f16(
            false, a, false, ones, (short)0, acc, false, false);
    }

    // Tail chunk c0 = 544: cols 544..559 live in elems 0..7 (per lane half);
    // elems 8..15 map to cols 560..575 which are always OOB -> constant 0.
    {
        v16h a;
#pragma unroll
        for (int j = 0; j < 8; j++) {
            int col  = 544 + hi + j;
            int colc = (col < CC) ? col : (CC - 1);  // clamped, always-legal load
            float x  = srow[colc] + s_lv[colc];
            x = (col == CC) ? (dust + s_lv[CC]) : x;
            x = (col >  CC) ? NEG_INF : x;           // col 559 only
            a[j]     = (_Float16)__expf(x);
            a[8 + j] = (_Float16)0.0f;               // cols 560..575: OOB
        }
        acc = __builtin_amdgcn_wmma_f32_16x16x32_f16(
            false, a, false, ones, (short)0, acc, false, false);
    }

    // C/D layout: VGPR i holds M=i (lanes 0-15) or M=8+i (lanes 16-31); all N equal.
    if ((lane & 15) == 0) {
        int rbase = (lane >> 4) * 8;
#pragma unroll
        for (int i = 0; i < 8; i++) {
            int n = n0 + rbase + i;
            logu[b * NN + n] = logmu[b * NN + n] - __logf(acc[i]);
        }
    }
}

// ---------------------------------------------------------------------------
// K2: column update. log_v[b,m] = log_nu - log( sum_n exp(aug[b,n,m]+log_u[b,n]) )
// 64 contiguous columns per block (coalesced 128B/wave loads), log_u staged
// to LDS via async b128 copies, 4 n-strips reduced through LDS.
// ---------------------------------------------------------------------------
__global__ void k_col(const float* __restrict__ scores,
                      const float* __restrict__ dustp,
                      const float* __restrict__ logu,
                      float* __restrict__ logv)
{
    int b  = blockIdx.x / 9;
    int m0 = (blockIdx.x % 9) * 64;
    int t  = threadIdx.x;
    int mc = t & 63;
    int g  = t >> 6;          // 4 n-strips
    int m  = m0 + mc;

    __shared__ __attribute__((aligned(16))) float s_lu[NN];
    __shared__ float s_part[256];
    stage_lds(s_lu, logu + (size_t)b * NN, NN, t, 256);
    __syncthreads();

    float dust = dustp[0];
    float s0 = 0.0f, s1 = 0.0f;
    if (m < CC) {
        const float* base = scores + (size_t)b * NN * CC + m;
        int n = g * (NN / 4);
        for (int i = 0; i < NN / 4; i += 2, n += 2) {
            __builtin_prefetch(&base[(size_t)(n + 32) * CC], 0, 0); // global_prefetch_b8
            s0 += __expf(base[(size_t)n * CC]       + s_lu[n]);
            s1 += __expf(base[(size_t)(n + 1) * CC] + s_lu[n + 1]);
        }
    } else if (m == CC) {
        int n = g * (NN / 4);
        for (int i = 0; i < NN / 4; i += 2, n += 2) {
            s0 += __expf(dust + s_lu[n]);
            s1 += __expf(dust + s_lu[n + 1]);
        }
    }
    s_part[t] = s0 + s1;
    __syncthreads();
    if (t < 64) {
        float tot = s_part[t] + s_part[t + 64] + s_part[t + 128] + s_part[t + 192];
        int mm = m0 + t;
        if (mm < MM) logv[b * MP + mm] = LOG_NU - __logf(tot);
    }
}

// ---------------------------------------------------------------------------
// K3: finalize.  full = exp(aug + log_u + log_v); soft = full[:, :CC];
// hard = argmax_m soft (first max); dust = full[:, CC].
// One wave per row; main loop divergence-free; wave32 shuffle argmax.
// ---------------------------------------------------------------------------
__global__ void k_final(const float* __restrict__ scores,
                        const float* __restrict__ dustp,
                        const float* __restrict__ logu,
                        const float* __restrict__ logv,
                        float* __restrict__ out)
{
    float* full = out;
    float* soft = out + (size_t)BB * NN * MM;
    float* hard = soft + (size_t)BB * NN * CC;
    float* dstw = hard + (size_t)BB * NN;

    int t    = threadIdx.x;
    int wave = t >> 5;
    int lane = t & 31;
    int b = blockIdx.x >> 9;                       // 512 blocks per batch
    int n = (int)((blockIdx.x & 511) * 8 + wave);  // 8 rows per block

    __shared__ __attribute__((aligned(16))) float s_lv[MP];
    stage_lds(s_lv, logv + (size_t)b * MP, MP, t, 256);
    __syncthreads();

    float lu   = logu[b * NN + n];
    float dust = dustp[0];
    const float* srow = scores + ((size_t)b * NN + n) * CC;
    float* frow = full + ((size_t)b * NN + n) * MM;
    float* prow = soft + ((size_t)b * NN + n) * CC;

    float bestv = -1.0f;
    int   besti = 0;
    // cols 0..543: all lanes in-bounds, no divergence.
    for (int c0 = 0; c0 < 544; c0 += 32) {
        int col = c0 + lane;
        float v = __expf(srow[col] + lu + s_lv[col]);
        frow[col] = v;
        prow[col] = v;
        if (v > bestv) { bestv = v; besti = col; }
    }
    // tail: cols 544..575 (lanes 0..13 scores, lane 14 dustbin, rest idle)
    {
        int col = 544 + lane;
        if (col < CC) {
            float v = __expf(srow[col] + lu + s_lv[col]);
            frow[col] = v;
            prow[col] = v;
            if (v > bestv) { bestv = v; besti = col; }
        } else if (col == CC) {
            float v = __expf(dust + lu + s_lv[CC]);
            frow[CC] = v;
            dstw[(size_t)b * NN + n] = v;
        }
    }
    // wave32 argmax, ties -> lowest index (matches jnp.argmax first-occurrence)
    for (int off = 16; off > 0; off >>= 1) {
        float ov = __shfl_xor(bestv, off, 32);
        int   oi = __shfl_xor(besti, off, 32);
        if (ov > bestv || (ov == bestv && oi < besti)) { bestv = ov; besti = oi; }
    }
    if (lane == 0) hard[(size_t)b * NN + n] = (float)besti;
}

// ---------------------------------------------------------------------------
extern "C" void kernel_launch(void* const* d_in, const int* in_sizes, int n_in,
                              void* d_out, int out_size, void* d_ws, size_t ws_size,
                              hipStream_t stream)
{
    const float* scores = (const float*)d_in[0];
    const int*   mask   = (const int*)d_in[1];
    const float* dust   = (const float*)d_in[2];
    float* out = (float*)d_out;

    float* ws    = (float*)d_ws;
    float* logmu = ws;                         // BB*NN
    float* logu  = ws + (size_t)BB * NN;       // BB*NN
    float* logv  = ws + (size_t)2 * BB * NN;   // BB*MP

    k_init<<<BB, 256, 0, stream>>>(mask, logmu, logv);
    for (int it = 0; it < ITERS; ++it) {
        k_row<<<BB * 32, 256, 0, stream>>>(scores, dust, logmu, logv, logu);
        k_col<<<BB * 9, 256, 0, stream>>>(scores, dust, logu, logv);
    }
    k_final<<<BB * NN / 8, 256, 0, stream>>>(scores, dust, logu, logv, out);
}
